// TransformerBlock_21784074126072
// MI455X (gfx1250) — matrix operand order repaired and checked by reading, not compiled, not run on hardware
//
#include <hip/hip_runtime.h>

typedef __attribute__((ext_vector_type(16))) __bf16 v16bf;
typedef __attribute__((ext_vector_type(8)))  __bf16 bf16x8;
typedef __attribute__((ext_vector_type(4)))  __bf16 bf16x4;
typedef __attribute__((ext_vector_type(8)))  float  v8f;

// Async global->LDS DMA (gfx1250, ASYNCcnt-tracked). Guarded: falls back to
// load+ds_store staging if the builtins are not exposed by this toolchain.
// ROCm clang types these builtins with generic `int4*` pointers (b128 = v4i).
#if defined(__AMDGCN__) &&                                             \
    __has_builtin(__builtin_amdgcn_global_load_async_to_lds_b128) &&   \
    __has_builtin(__builtin_amdgcn_s_wait_asynccnt)
#define USE_ASYNC_COPY 1
#else
#define USE_ASYNC_COPY 0
#endif

typedef int v4i_vs __attribute__((vector_size(16)));
#define GPTR(p) ((v4i_vs*)(void*)(p))
#define LPTR(p) ((v4i_vs*)(void*)(p))

// ---------------------------------------------------------------------------
// Shapes (fixed by the reference): B=32, S=1024, D=256, F=512, H=8, d=32
// ---------------------------------------------------------------------------

enum {
  EPI_QH     = 0,  // +bias, + hidden[b,h,s,j] remap, store bf16 [32768,256]
  EPI_BIAS   = 1,  // +bias, store bf16 [32768,256]
  EPI_VT     = 2,  // +bias, store bf16 transposed per head: vT[b*8+h][j][s]
  EPI_SCORES = 3,  // *scale, store fp32 attn[(b*8+h)][q][k]
  EPI_CTX    = 4,  // store bf16 ctx[b*1024+q][h*32+n]
  EPI_OPROJ  = 5,  // 2*(acc+bias), store fp32 [32768,256]
  EPI_FF1    = 6,  // relu(acc+bias), store bf16 [32768,512]
  EPI_FF2    = 7   // acc+bias+resid, store fp32 [32768,256] (in-place resid)
};

struct GemmP {
  const void*   A;     // bf16 (or fp32 when AF32)
  const __bf16* B;     // [N,K] bf16
  const float*  bias;
  const float*  aux0;  // hidden (EPI_QH) / residual (EPI_FF2)
  void*         out;
  float         scale;
  int M, N, K, lda, ldb;
  long long sA0, sA1, sB0, sB1;  // batch offsets: off = zb*s0 + zh*s1 (z = zb*8+zh)
};

template <int EPI>
__device__ __forceinline__ void epi_store(const GemmP& p, int zb, int zh,
                                          int row, int col, float acc) {
  if (row >= p.M || col >= p.N) return;
  if constexpr (EPI == EPI_QH) {
    float v = acc + p.bias[col];
    int b = row >> 10, s = row & 1023, hh = col >> 5, j = col & 31;
    v += p.aux0[((((size_t)b * 8 + hh) * 1024 + s) * 32) + j];
    ((__bf16*)p.out)[(size_t)row * 256 + col] = (__bf16)v;
  } else if constexpr (EPI == EPI_BIAS) {
    ((__bf16*)p.out)[(size_t)row * 256 + col] = (__bf16)(acc + p.bias[col]);
  } else if constexpr (EPI == EPI_VT) {
    float v = acc + p.bias[col];
    int b = row >> 10, s = row & 1023, hh = col >> 5, j = col & 31;
    ((__bf16*)p.out)[(((size_t)(b * 8 + hh) * 32 + j) * 1024) + s] = (__bf16)v;
  } else if constexpr (EPI == EPI_SCORES) {
    size_t base = ((size_t)(zb * 8 + zh)) << 20;  // *1024*1024
    ((float*)p.out)[base + (size_t)row * 1024 + col] = acc * p.scale;
  } else if constexpr (EPI == EPI_CTX) {
    ((__bf16*)p.out)[((size_t)zb * 1024 + row) * 256 + zh * 32 + col] = (__bf16)acc;
  } else if constexpr (EPI == EPI_OPROJ) {
    ((float*)p.out)[(size_t)row * 256 + col] = 2.f * (acc + p.bias[col]);
  } else if constexpr (EPI == EPI_FF1) {
    ((__bf16*)p.out)[(size_t)row * 512 + col] = (__bf16)fmaxf(acc + p.bias[col], 0.f);
  } else {  // EPI_FF2
    ((float*)p.out)[(size_t)row * 256 + col] =
        acc + p.bias[col] + p.aux0[(size_t)row * 256 + col];
  }
}

__device__ __forceinline__ v16bf ld_frag(const __bf16* base, int second_off) {
  bf16x8 lo = *(const bf16x8*)base;
  bf16x8 hi = *(const bf16x8*)(base + second_off);
  return __builtin_shufflevector(lo, hi, 0, 1, 2, 3, 4, 5, 6, 7, 8, 9, 10, 11,
                                 12, 13, 14, 15);
}

// ---------------------------------------------------------------------------
// Tiled WMMA GEMM:  C[M,N] = A[M,K] * B[N,K]^T   (bf16 inputs, f32 accum)
// 256 threads = 8 wave32s, 128x64 block tile, BK=32, double-buffered LDS with
// async global->LDS DMA; each wave computes a 32x32 strip = 4 WMMAs per step.
// ---------------------------------------------------------------------------
template <int EPI, bool AF32>
__global__ __launch_bounds__(256) void wmma_gemm_kernel(GemmP p) {
  constexpr int BM = 128, BN = 64, BK = 32, LDS_W = BK + 8;
  __shared__ __align__(16) __bf16 As[2][BM][LDS_W];
  __shared__ __align__(16) __bf16 Bs[2][BN][LDS_W];

  const int tid  = threadIdx.x;
  const int lane = tid & 31;
  const int wid  = tid >> 5;
  const int wrow = wid >> 1;       // 0..3  (32-row strip)
  const int wcol = wid & 1;        // 0..1  (32-col strip)
  const int half = lane >> 4;      // wave32 half, per WMMA VGPR layout
  const int l16  = lane & 15;

  const int z  = blockIdx.z;
  const int zb = z >> 3, zh = z & 7;
  const int m0 = blockIdx.x * BM;
  const int n0 = blockIdx.y * BN;

  const size_t offA = (size_t)zb * p.sA0 + (size_t)zh * p.sA1;
  const size_t offB = (size_t)zb * p.sB0 + (size_t)zh * p.sB1;
  const __bf16* Ab = (const __bf16*)p.A + offA;
  const float*  Af = (const float*)p.A + offA;
  const __bf16* Bb = p.B + offB;

  // staging map: thread t covers rows sr and sr+64 of A, row sr of B,
  // 8-element column group scol; one 16B transfer each.
  const int sr   = tid >> 2;        // 0..63
  const int scol = (tid & 3) * 8;   // 0,8,16,24
  int ra0 = m0 + sr;      if (ra0 >= p.M) ra0 = p.M - 1;
  int ra1 = m0 + sr + 64; if (ra1 >= p.M) ra1 = p.M - 1;
  int rb  = n0 + sr;      if (rb  >= p.N) rb  = p.N - 1;

  auto stage = [&](int buf, int k0) {
#if USE_ASYNC_COPY
    if constexpr (AF32) {  // fp32 attn probs -> bf16 while staging (sync path)
      const float* s0 = Af + (size_t)ra0 * p.lda + (k0 + scol);
      const float* s1 = Af + (size_t)ra1 * p.lda + (k0 + scol);
      bf16x8 v0, v1;
#pragma unroll
      for (int i = 0; i < 8; ++i) { v0[i] = (__bf16)s0[i]; v1[i] = (__bf16)s1[i]; }
      *(bf16x8*)&As[buf][sr][scol]      = v0;
      *(bf16x8*)&As[buf][sr + 64][scol] = v1;
    } else {
      __builtin_amdgcn_global_load_async_to_lds_b128(
          GPTR(Ab + (size_t)ra0 * p.lda + (k0 + scol)),
          LPTR(&As[buf][sr][scol]), 0, 0);
      __builtin_amdgcn_global_load_async_to_lds_b128(
          GPTR(Ab + (size_t)ra1 * p.lda + (k0 + scol)),
          LPTR(&As[buf][sr + 64][scol]), 0, 0);
    }
    __builtin_amdgcn_global_load_async_to_lds_b128(
        GPTR(Bb + (size_t)rb * p.ldb + (k0 + scol)),
        LPTR(&Bs[buf][sr][scol]), 0, 0);
#else
    if constexpr (AF32) {
      const float* s0 = Af + (size_t)ra0 * p.lda + (k0 + scol);
      const float* s1 = Af + (size_t)ra1 * p.lda + (k0 + scol);
      bf16x8 v0, v1;
#pragma unroll
      for (int i = 0; i < 8; ++i) { v0[i] = (__bf16)s0[i]; v1[i] = (__bf16)s1[i]; }
      *(bf16x8*)&As[buf][sr][scol]      = v0;
      *(bf16x8*)&As[buf][sr + 64][scol] = v1;
    } else {
      *(bf16x8*)&As[buf][sr][scol] =
          *(const bf16x8*)(Ab + (size_t)ra0 * p.lda + (k0 + scol));
      *(bf16x8*)&As[buf][sr + 64][scol] =
          *(const bf16x8*)(Ab + (size_t)ra1 * p.lda + (k0 + scol));
    }
    *(bf16x8*)&Bs[buf][sr][scol] =
        *(const bf16x8*)(Bb + (size_t)rb * p.ldb + (k0 + scol));
#endif
  };

  v8f acc00 = {}, acc01 = {}, acc10 = {}, acc11 = {};

  stage(0, 0);
  for (int k0 = 0, it = 0; k0 < p.K; k0 += BK, ++it) {
    const int buf = it & 1;
#if USE_ASYNC_COPY
    __builtin_amdgcn_s_wait_asynccnt(0);  // our DMA into `buf` is complete
#endif
    __syncthreads();                       // everyone's staging/reads complete
    if (k0 + BK < p.K) stage(buf ^ 1, k0 + BK);  // overlap DMA with compute

    // A frags 16x32 (lanes 0-15: K{0..7,16..23}; lanes 16-31: K{8..15,24..31})
    const int kA = half ? 8 : 0;
    v16bf a0 = ld_frag(&As[buf][wrow * 32 + l16][kA], 16);
    v16bf a1 = ld_frag(&As[buf][wrow * 32 + 16 + l16][kA], 16);
    // B frags 32x16 (lanes 0-15: K{0..7,16..23}; lanes 16-31: K{8..15,24..31}), col = l16
    const int kB = half ? 8 : 0;
    v16bf b0 = ld_frag(&Bs[buf][wcol * 32 + l16][kB], 16);
    v16bf b1 = ld_frag(&Bs[buf][wcol * 32 + 16 + l16][kB], 16);

    acc00 = __builtin_amdgcn_wmma_f32_16x16x32_bf16(false, a0, false, b0,
                                                    (short)0, acc00, false, false);
    acc01 = __builtin_amdgcn_wmma_f32_16x16x32_bf16(false, a0, false, b1,
                                                    (short)0, acc01, false, false);
    acc10 = __builtin_amdgcn_wmma_f32_16x16x32_bf16(false, a1, false, b0,
                                                    (short)0, acc10, false, false);
    acc11 = __builtin_amdgcn_wmma_f32_16x16x32_bf16(false, a1, false, b1,
                                                    (short)0, acc11, false, false);
  }

  // C/D layout: VGPR i -> row = base + half*8 + i, col = l16 (+16 for tile 1)
  const int r0 = m0 + wrow * 32 + half * 8;
  const int r1 = r0 + 16;
  const int c0 = n0 + wcol * 32 + l16;
#pragma unroll
  for (int i = 0; i < 8; ++i) {
    epi_store<EPI>(p, zb, zh, r0 + i, c0,      acc00[i]);
    epi_store<EPI>(p, zb, zh, r0 + i, c0 + 16, acc01[i]);
    epi_store<EPI>(p, zb, zh, r1 + i, c0,      acc10[i]);
    epi_store<EPI>(p, zb, zh, r1 + i, c0 + 16, acc11[i]);
  }
}

// ---------------------------------------------------------------------------
// fp32 -> bf16 pack (vectorized x4)
// ---------------------------------------------------------------------------
__global__ void cvt_f32_bf16_kernel(const float* __restrict__ s,
                                    __bf16* __restrict__ d, int n4) {
  int i = blockIdx.x * 256 + threadIdx.x;
  if (i < n4) {
    float4 v = ((const float4*)s)[i];
    bf16x4 o = {(__bf16)v.x, (__bf16)v.y, (__bf16)v.z, (__bf16)v.w};
    ((bf16x4*)d)[i] = o;
  }
}

// ---------------------------------------------------------------------------
// Row softmax, in-place on fp32 attn rows of length 1024; one wave32 per row.
// ---------------------------------------------------------------------------
__global__ __launch_bounds__(256) void softmax_rows_kernel(float* __restrict__ attn) {
  const int row  = blockIdx.x * 8 + (threadIdx.x >> 5);
  const int lane = threadIdx.x & 31;
  float4* rp = (float4*)(attn + (size_t)row * 1024);
  float4 v[8];
  float mx = -3.0e38f;
#pragma unroll
  for (int i = 0; i < 8; ++i) {
    v[i] = rp[lane * 8 + i];
    mx = fmaxf(mx, fmaxf(fmaxf(v[i].x, v[i].y), fmaxf(v[i].z, v[i].w)));
  }
#pragma unroll
  for (int off = 16; off > 0; off >>= 1) mx = fmaxf(mx, __shfl_xor(mx, off, 32));
  float sum = 0.f;
#pragma unroll
  for (int i = 0; i < 8; ++i) {
    v[i].x = __expf(v[i].x - mx);
    v[i].y = __expf(v[i].y - mx);
    v[i].z = __expf(v[i].z - mx);
    v[i].w = __expf(v[i].w - mx);
    sum += v[i].x + v[i].y + v[i].z + v[i].w;
  }
#pragma unroll
  for (int off = 16; off > 0; off >>= 1) sum += __shfl_xor(sum, off, 32);
  const float inv = 1.f / sum;
#pragma unroll
  for (int i = 0; i < 8; ++i) {
    v[i].x *= inv; v[i].y *= inv; v[i].z *= inv; v[i].w *= inv;
    rp[lane * 8 + i] = v[i];
  }
}

// ---------------------------------------------------------------------------
// Row LayerNorm over D=256, one wave32 per row. Outputs bf16 (for FFN input).
// ---------------------------------------------------------------------------
__global__ __launch_bounds__(256) void rowln_bf16_kernel(
    const float* __restrict__ h, const float* __restrict__ g,
    const float* __restrict__ be, __bf16* __restrict__ out) {
  const int row  = blockIdx.x * 8 + (threadIdx.x >> 5);
  const int lane = threadIdx.x & 31;
  const float4* rp = (const float4*)(h + (size_t)row * 256);
  float4 a = rp[lane * 2], b = rp[lane * 2 + 1];
  float s = a.x + a.y + a.z + a.w + b.x + b.y + b.z + b.w;
#pragma unroll
  for (int off = 16; off > 0; off >>= 1) s += __shfl_xor(s, off, 32);
  const float mu = s * (1.f / 256.f);
  float e[8] = {a.x - mu, a.y - mu, a.z - mu, a.w - mu,
                b.x - mu, b.y - mu, b.z - mu, b.w - mu};
  float q = 0.f;
#pragma unroll
  for (int i = 0; i < 8; ++i) q += e[i] * e[i];
#pragma unroll
  for (int off = 16; off > 0; off >>= 1) q += __shfl_xor(q, off, 32);
  const float r = rsqrtf(q * (1.f / 256.f) + 1e-5f);
  const int c0 = lane * 8;
  bf16x8 o;
#pragma unroll
  for (int i = 0; i < 8; ++i) o[i] = (__bf16)(e[i] * r * g[c0 + i] + be[c0 + i]);
  *(bf16x8*)(out + (size_t)row * 256 + c0) = o;
}

// ---------------------------------------------------------------------------
// Final: out = h + LayerNorm(h)  (fp32 output rows)
// ---------------------------------------------------------------------------
__global__ __launch_bounds__(256) void final_addln_kernel(
    const float* __restrict__ h, const float* __restrict__ g,
    const float* __restrict__ be, float* __restrict__ out) {
  const int row  = blockIdx.x * 8 + (threadIdx.x >> 5);
  const int lane = threadIdx.x & 31;
  const float4* rp = (const float4*)(h + (size_t)row * 256);
  float4 a = rp[lane * 2], b = rp[lane * 2 + 1];
  float s = a.x + a.y + a.z + a.w + b.x + b.y + b.z + b.w;
#pragma unroll
  for (int off = 16; off > 0; off >>= 1) s += __shfl_xor(s, off, 32);
  const float mu = s * (1.f / 256.f);
  float xv[8] = {a.x, a.y, a.z, a.w, b.x, b.y, b.z, b.w};
  float q = 0.f;
#pragma unroll
  for (int i = 0; i < 8; ++i) { float e = xv[i] - mu; q += e * e; }
#pragma unroll
  for (int off = 16; off > 0; off >>= 1) q += __shfl_xor(q, off, 32);
  const float r = rsqrtf(q * (1.f / 256.f) + 1e-5f);
  const int c0 = lane * 8;
  float4 o0, o1;
#pragma unroll
  for (int i = 0; i < 8; ++i) {
    float ln = (xv[i] - mu) * r * g[c0 + i] + be[c0 + i];
    ((i < 4) ? (&o0.x)[i] : (&o1.x)[i - 4]) = xv[i] + ln;
  }
  float4* wp = (float4*)(out + (size_t)row * 256 + c0);
  wp[0] = o0;
  wp[1] = o1;
}

// ---------------------------------------------------------------------------
// Orchestration
// ---------------------------------------------------------------------------
extern "C" void kernel_launch(void* const* d_in, const int* in_sizes, int n_in,
                              void* d_out, int out_size, void* d_ws,
                              size_t ws_size, hipStream_t stream) {
  (void)in_sizes; (void)n_in; (void)out_size; (void)ws_size;
  const float* x      = (const float*)d_in[0];
  const float* hidden = (const float*)d_in[1];
  const float* Wq = (const float*)d_in[2];  const float* bq = (const float*)d_in[3];
  const float* Wk = (const float*)d_in[4];  const float* bk = (const float*)d_in[5];
  const float* Wv = (const float*)d_in[6];  const float* bv = (const float*)d_in[7];
  const float* Wo = (const float*)d_in[8];  const float* bo = (const float*)d_in[9];
  const float* g1 = (const float*)d_in[10]; const float* be1 = (const float*)d_in[11];
  const float* g2 = (const float*)d_in[12]; const float* be2 = (const float*)d_in[13];
  const float* W1 = (const float*)d_in[14]; const float* b1 = (const float*)d_in[15];
  const float* W2 = (const float*)d_in[16]; const float* b2 = (const float*)d_in[17];

  constexpr size_t D = 256, F = 512;
  constexpr size_t MS = 32768;  // B*S
  float* outF  = (float*)d_out;
  float* attnO = outF + MS * D;  // attn region of the tuple output

  char* ws = (char*)d_ws;
  size_t off = 0;
  auto take = [&](size_t bytes) -> char* {
    char* r = ws + off;
    off = (off + bytes + 255) & ~(size_t)255;
    return r;
  };
  __bf16* xb  = (__bf16*)take(MS * D * 2);
  __bf16* qh  = (__bf16*)take(MS * D * 2);
  __bf16* kb  = (__bf16*)take(MS * D * 2);
  __bf16* vT  = (__bf16*)take(MS * D * 2);
  __bf16* ctx = (__bf16*)take(MS * D * 2);
  float*  hb  = (float*) take(MS * D * 4);
  __bf16* hn  = (__bf16*)take(MS * D * 2);
  __bf16* ff1 = (__bf16*)take(MS * F * 2);
  __bf16* wqb = (__bf16*)take(D * D * 2);
  __bf16* wkb = (__bf16*)take(D * D * 2);
  __bf16* wvb = (__bf16*)take(D * D * 2);
  __bf16* wob = (__bf16*)take(D * D * 2);
  __bf16* w1b = (__bf16*)take(F * D * 2);
  __bf16* w2b = (__bf16*)take(D * F * 2);

  auto cvt = [&](const float* s, __bf16* d, size_t n) {
    int n4 = (int)(n / 4);
    cvt_f32_bf16_kernel<<<(n4 + 255) / 256, 256, 0, stream>>>(s, d, n4);
  };
  cvt(x, xb, MS * D);
  cvt(Wq, wqb, D * D); cvt(Wk, wkb, D * D); cvt(Wv, wvb, D * D);
  cvt(Wo, wob, D * D); cvt(W1, w1b, F * D); cvt(W2, w2b, D * F);

  GemmP p{};

  // Q projection, epilogue fuses bias + hidden (since (q+hidden)k^T)
  p = {}; p.A = xb; p.B = wqb; p.bias = bq; p.aux0 = hidden; p.out = qh;
  p.M = 32768; p.N = 256; p.K = 256; p.lda = 256; p.ldb = 256;
  wmma_gemm_kernel<EPI_QH, false><<<dim3(256, 4, 1), 256, 0, stream>>>(p);

  // K projection
  p = {}; p.A = xb; p.B = wkb; p.bias = bk; p.out = kb;
  p.M = 32768; p.N = 256; p.K = 256; p.lda = 256; p.ldb = 256;
  wmma_gemm_kernel<EPI_BIAS, false><<<dim3(256, 4, 1), 256, 0, stream>>>(p);

  // V projection, stored per-head transposed vT[b*8+h][j][s]
  p = {}; p.A = xb; p.B = wvb; p.bias = bv; p.out = vT;
  p.M = 32768; p.N = 256; p.K = 256; p.lda = 256; p.ldb = 256;
  wmma_gemm_kernel<EPI_VT, false><<<dim3(256, 4, 1), 256, 0, stream>>>(p);

  // Scores: per (b,h): [1024,32] @ [1024,32]^T * scale -> fp32 attn (d_out)
  p = {}; p.A = qh; p.B = kb; p.out = attnO; p.scale = 0.17677669529663687f;
  p.M = 1024; p.N = 1024; p.K = 32; p.lda = 256; p.ldb = 256;
  p.sA0 = 1024 * 256; p.sA1 = 32; p.sB0 = 1024 * 256; p.sB1 = 32;
  wmma_gemm_kernel<EPI_SCORES, false><<<dim3(8, 16, 256), 256, 0, stream>>>(p);

  // Softmax rows in-place on attn (B*H*S = 262144 rows of 1024)
  softmax_rows_kernel<<<dim3(32768), 256, 0, stream>>>(attnO);

  // ctx: per (b,h): attn[1024,1024] @ vT[32,1024]^T ; A is fp32, bf16ed on stage
  p = {}; p.A = attnO; p.B = vT; p.out = ctx;
  p.M = 1024; p.N = 32; p.K = 1024; p.lda = 1024; p.ldb = 1024;
  p.sA0 = 8LL * 1024 * 1024; p.sA1 = 1024 * 1024;
  p.sB0 = 8LL * 32 * 1024;   p.sB1 = 32 * 1024;
  wmma_gemm_kernel<EPI_CTX, true><<<dim3(8, 1, 256), 256, 0, stream>>>(p);

  // Output projection; h = 2*(ctx@Wo^T + bo)
  p = {}; p.A = ctx; p.B = wob; p.bias = bo; p.out = hb;
  p.M = 32768; p.N = 256; p.K = 256; p.lda = 256; p.ldb = 256;
  wmma_gemm_kernel<EPI_OPROJ, false><<<dim3(256, 4, 1), 256, 0, stream>>>(p);

  // LayerNorm1 -> bf16
  rowln_bf16_kernel<<<dim3(4096), 256, 0, stream>>>(hb, g1, be1, hn);

  // FFN1: relu(hn @ W1^T + b1) -> bf16 [32768,512]
  p = {}; p.A = hn; p.B = w1b; p.bias = b1; p.out = ff1;
  p.M = 32768; p.N = 512; p.K = 256; p.lda = 256; p.ldb = 256;
  wmma_gemm_kernel<EPI_FF1, false><<<dim3(256, 8, 1), 256, 0, stream>>>(p);

  // FFN2 + residual: hb = hb + ff1 @ W2^T + b2  (in-place, 1:1 element map)
  p = {}; p.A = ff1; p.B = w2b; p.bias = b2; p.aux0 = hb; p.out = hb;
  p.M = 32768; p.N = 256; p.K = 512; p.lda = 512; p.ldb = 512;
  wmma_gemm_kernel<EPI_FF2, false><<<dim3(256, 4, 1), 256, 0, stream>>>(p);

  // Final: out = h + LayerNorm(h)
  final_addln_kernel<<<dim3(4096), 256, 0, stream>>>(hb, g2, be2, outF);
}
